// MambaBlock_34832184770990
// MI455X (gfx1250) — matrix operand order repaired and checked by reading, not compiled, not run on hardware
//
#include <hip/hip_runtime.h>
#include <math.h>

typedef __attribute__((ext_vector_type(16))) __bf16 v16bf;
typedef __attribute__((ext_vector_type(8)))  __bf16 v8bf;
typedef __attribute__((ext_vector_type(4)))  __bf16 v4bf;
typedef __attribute__((ext_vector_type(8)))  float  v8f;

#define D_MODEL 512
#define D_INNER 1024
#define D_STATE 16
#define DT_RANK 32
#define SEQ_L   2048
#define BATCH   8
#define BLROWS  (BATCH * SEQ_L)   // 16384

// ---------------------------------------------------------------------------
// Tiled bf16 WMMA GEMM: C[M,N] = epilogue(A[M,K](row stride lda) @ W[K,N])
// Block tile 256x64, BK=32, 256 threads = 8 waves, each wave a 32x64 strip
// (2 M-subtiles x 4 N-subtiles = 8 WMMAs per K-step, B frags reused 2x).
// Double-buffered LDS pipeline: global loads of tile k+1 issue before the
// WMMAs of tile k; cvt+LDS-store (and the loadcnt wait) land after them.
// LDS: As[m][k] row-major (A frag = two contiguous b128 runs per lane),
//      Bs[n][k] transposed  (B frag = one contiguous 32B run per lane).
// MODE 0: raw   MODE 2: softplus(x + bias)   MODE 3: gelu(x + bias) + resid
// ---------------------------------------------------------------------------
template <int MODE>
__global__ __launch_bounds__(256)
void gemm_wmma_kernel(const float* __restrict__ A, int lda,
                      const float* __restrict__ W,
                      const float* __restrict__ bias,
                      const float* __restrict__ resid,
                      float* __restrict__ C,
                      int M, int N, int K)
{
    __shared__ __bf16 As[2][256][32];   // 2 x 16 KB, [m][k]
    __shared__ __bf16 Bs[2][64][32];    // 2 x  4 KB, [n][k] (transposed)

    const int tid    = threadIdx.x;
    const int w      = tid >> 5;          // wave id 0..7
    const int lane   = tid & 31;
    const int lane16 = lane & 15;
    const int half   = (lane >> 4) & 1;
    const int m0     = blockIdx.y * 256;
    const int n0     = blockIdx.x * 64;

    v8f acc[2][4] = {};

    float4 aReg[8];
    float4 bReg[2];

    // Per-thread staging coordinates (A: 2048 float4 slots, B: 512 slots).
    auto loadRegs = [&](int k0) {
#pragma unroll
        for (int i = 0; i < 8; ++i) {
            int idx = tid + i * 256;
            int m   = idx >> 3;
            int kk  = (idx & 7) * 4;
            aReg[i] = *(const float4*)&A[(size_t)(m0 + m) * lda + (k0 + kk)];
        }
#pragma unroll
        for (int i = 0; i < 2; ++i) {
            int idx = tid + i * 256;
            int kk  = idx >> 4;
            int n   = (idx & 15) * 4;
            bReg[i] = *(const float4*)&W[(size_t)(k0 + kk) * N + (n0 + n)];
        }
    };
    auto storeLds = [&](int sel) {
#pragma unroll
        for (int i = 0; i < 8; ++i) {
            int idx = tid + i * 256;
            int m   = idx >> 3;
            int kk  = (idx & 7) * 4;
            v4bf p; p[0] = (__bf16)aReg[i].x; p[1] = (__bf16)aReg[i].y;
                    p[2] = (__bf16)aReg[i].z; p[3] = (__bf16)aReg[i].w;
            *(v4bf*)&As[sel][m][kk] = p;
        }
#pragma unroll
        for (int i = 0; i < 2; ++i) {
            int idx = tid + i * 256;
            int kk  = idx >> 4;
            int n   = (idx & 15) * 4;
            Bs[sel][n + 0][kk] = (__bf16)bReg[i].x;
            Bs[sel][n + 1][kk] = (__bf16)bReg[i].y;
            Bs[sel][n + 2][kk] = (__bf16)bReg[i].z;
            Bs[sel][n + 3][kk] = (__bf16)bReg[i].w;
        }
    };

    // ---- pipeline prologue: stage tile 0 ----
    loadRegs(0);
    storeLds(0);
    __syncthreads();

    int cur = 0;
    for (int k0 = 0; k0 < K; k0 += 32) {
        const bool more = (k0 + 32) < K;
        if (more) loadRegs(k0 + 32);    // fire next tile's global loads now

        // ---- A fragments: two contiguous v8bf runs per lane, shuffled ----
        const int ar0 = w * 32 + lane16;
        const int ar1 = ar0 + 16;
        v8bf a0lo = *(const v8bf*)&As[cur][ar0][16 * half];
        v8bf a0hi = *(const v8bf*)&As[cur][ar0][16 * half + 8];
        v8bf a1lo = *(const v8bf*)&As[cur][ar1][16 * half];
        v8bf a1hi = *(const v8bf*)&As[cur][ar1][16 * half + 8];
        v16bf afrag0 = __builtin_shufflevector(a0lo, a0hi,
            0,1,2,3,4,5,6,7,8,9,10,11,12,13,14,15);
        v16bf afrag1 = __builtin_shufflevector(a1lo, a1hi,
            0,1,2,3,4,5,6,7,8,9,10,11,12,13,14,15);

        // ---- 4 B fragments (one 32B contiguous LDS read each), 8 WMMAs ----
#pragma unroll
        for (int t = 0; t < 4; ++t) {
            v16bf bfrag = *(const v16bf*)&Bs[cur][t * 16 + lane16][16 * half];
            acc[0][t] = __builtin_amdgcn_wmma_f32_16x16x32_bf16(
                false, afrag0, false, bfrag, (short)0, acc[0][t], false, false);
            acc[1][t] = __builtin_amdgcn_wmma_f32_16x16x32_bf16(
                false, afrag1, false, bfrag, (short)0, acc[1][t], false, false);
        }

        if (more) storeLds(cur ^ 1);    // waits on loads here, after the WMMAs
        __syncthreads();
        cur ^= 1;
    }

    // ---- epilogue + store (C/D layout: VGPR r -> M = r + 8*half) ----
#pragma unroll
    for (int s = 0; s < 2; ++s) {
#pragma unroll
        for (int t = 0; t < 4; ++t) {
#pragma unroll
            for (int r = 0; r < 8; ++r) {
                int grow = m0 + w * 32 + s * 16 + r + 8 * half;
                int gcol = n0 + t * 16 + lane16;
                float v = acc[s][t][r];
                if (MODE >= 2) v += bias[gcol];
                if (MODE == 2) v = (v > 20.0f) ? v : log1pf(__expf(v));
                if (MODE == 3) {
                    float g = 0.5f * v * (1.0f + erff(v * 0.70710678118654752f));
                    v = g + resid[(size_t)grow * N + gcol];
                }
                C[(size_t)grow * N + gcol] = v;
            }
        }
    }
}

// ---------------------------------------------------------------------------
// Depthwise causal conv (k=4) + bias + SiLU.  u source = xz[:, :, 0:1024].
// ---------------------------------------------------------------------------
__global__ void conv_silu_kernel(const float* __restrict__ xz,
                                 const float* __restrict__ cw,
                                 const float* __restrict__ cb,
                                 float* __restrict__ u, int total)
{
    int idx = blockIdx.x * blockDim.x + threadIdx.x;
    if (idx >= total) return;
    int c  = idx & (D_INNER - 1);
    int bl = idx >> 10;
    int l  = bl & (SEQ_L - 1);
    float acc = cb[c];
#pragma unroll
    for (int k = 0; k < 4; ++k) {
        int ls = l - 3 + k;
        if (ls >= 0)
            acc += cw[k * D_INNER + c] * xz[(size_t)(bl - (3 - k)) * (2 * D_INNER) + c];
    }
    u[(size_t)bl * D_INNER + c] = acc / (1.0f + __expf(-acc));  // silu
}

// ---------------------------------------------------------------------------
// Selective scan: one thread per (batch, inner channel); N=16 state in regs.
// Fuses the u*Dp skip and the silu(z) output gate.
// ---------------------------------------------------------------------------
__global__ void scan_kernel(const float* __restrict__ u,
                            const float* __restrict__ delta,
                            const float* __restrict__ dbc,
                            const float* __restrict__ xz,
                            const float* __restrict__ A_log,
                            const float* __restrict__ Dp,
                            float* __restrict__ y)
{
    int idx = blockIdx.x * blockDim.x + threadIdx.x;
    if (idx >= BATCH * D_INNER) return;
    int d = idx & (D_INNER - 1);
    int b = idx >> 10;

    float Arow[D_STATE], h[D_STATE];
#pragma unroll
    for (int n = 0; n < D_STATE; ++n) {
        Arow[n] = -__expf(A_log[d * D_STATE + n]);
        h[n] = 0.0f;
    }
    const float Dd = Dp[d];

    for (int t = 0; t < SEQ_L; ++t) {
        size_t bl = (size_t)b * SEQ_L + t;
        float dlt = delta[bl * D_INNER + d];
        float ut  = u[bl * D_INNER + d];
        float du  = dlt * ut;
        const float* bc = dbc + bl * (DT_RANK + 2 * D_STATE);
        float yv = 0.0f;
#pragma unroll
        for (int n = 0; n < D_STATE; ++n) {
            float dA = __expf(dlt * Arow[n]);
            h[n] = dA * h[n] + du * bc[DT_RANK + n];
            yv += h[n] * bc[DT_RANK + D_STATE + n];
        }
        yv += ut * Dd;
        float zt = xz[bl * (2 * D_INNER) + D_INNER + d];
        yv *= zt / (1.0f + __expf(-zt));    // * silu(z)
        y[bl * D_INNER + d] = yv;
    }
}

// ---------------------------------------------------------------------------
// LayerNorm over last dim (512), one block per row.
// ---------------------------------------------------------------------------
__global__ __launch_bounds__(256)
void ln_kernel(const float* __restrict__ h, const float* __restrict__ g,
               const float* __restrict__ bta, float* __restrict__ o)
{
    __shared__ float s1[256], s2[256];
    const int row = blockIdx.x;
    const float* hr = h + (size_t)row * D_MODEL;
    float a = 0.0f, b2 = 0.0f;
    for (int i = threadIdx.x; i < D_MODEL; i += 256) {
        float v = hr[i];
        a += v; b2 += v * v;
    }
    s1[threadIdx.x] = a; s2[threadIdx.x] = b2;
    __syncthreads();
    for (int s = 128; s > 0; s >>= 1) {
        if (threadIdx.x < s) {
            s1[threadIdx.x] += s1[threadIdx.x + s];
            s2[threadIdx.x] += s2[threadIdx.x + s];
        }
        __syncthreads();
    }
    float mu  = s1[0] * (1.0f / D_MODEL);
    float var = s2[0] * (1.0f / D_MODEL) - mu * mu;
    float rs  = rsqrtf(var + 1e-5f);
    for (int i = threadIdx.x; i < D_MODEL; i += 256)
        o[(size_t)row * D_MODEL + i] = (hr[i] - mu) * rs * g[i] + bta[i];
}

// ---------------------------------------------------------------------------
extern "C" void kernel_launch(void* const* d_in, const int* in_sizes, int n_in,
                              void* d_out, int out_size, void* d_ws, size_t ws_size,
                              hipStream_t stream)
{
    const float* x    = (const float*)d_in[0];   // [8,2048,512]
    const float* ipw  = (const float*)d_in[1];   // [512,2048]
    const float* cw   = (const float*)d_in[2];   // [4,1024]
    const float* cb   = (const float*)d_in[3];   // [1024]
    const float* xpw  = (const float*)d_in[4];   // [1024,64]
    const float* dtw  = (const float*)d_in[5];   // [32,1024]
    const float* dtb  = (const float*)d_in[6];   // [1024]
    const float* alog = (const float*)d_in[7];   // [1024,16]
    const float* Dp   = (const float*)d_in[8];   // [1024]
    const float* opw  = (const float*)d_in[9];   // [1024,512]
    const float* lng  = (const float*)d_in[10];  // [512]
    const float* lnb  = (const float*)d_in[11];  // [512]
    const float* linw = (const float*)d_in[12];  // [512,512]
    const float* linb = (const float*)d_in[13];  // [512]
    float* out = (float*)d_out;

    float* ws    = (float*)d_ws;
    float* xz    = ws;                                   // [16384, 2048]
    float* u     = xz    + (size_t)BLROWS * 2048;        // [16384, 1024]
    float* dbc   = u     + (size_t)BLROWS * 1024;        // [16384, 64]
    float* delta = dbc   + (size_t)BLROWS * 64;          // [16384, 1024]
    float* y     = delta + (size_t)BLROWS * 1024;        // [16384, 1024]
    float* h     = y     + (size_t)BLROWS * 1024;        // [16384, 512]
    float* h2    = h     + (size_t)BLROWS * 512;         // [16384, 512]

    dim3 blk(256);
    const int gy = BLROWS / 256;   // 64

    // 1. in_proj: xz = x @ in_proj_w            [16384,2048] K=512
    gemm_wmma_kernel<0><<<dim3(2048 / 64, gy), blk, 0, stream>>>(
        x, 512, ipw, nullptr, nullptr, xz, BLROWS, 2048, 512);

    // 2. depthwise conv + silu -> u
    int n1 = BLROWS * D_INNER;
    conv_silu_kernel<<<(n1 + 255) / 256, blk, 0, stream>>>(xz, cw, cb, u, n1);

    // 3. x_proj: dbc = u @ x_proj_w             [16384,64] K=1024
    gemm_wmma_kernel<0><<<dim3(64 / 64, gy), blk, 0, stream>>>(
        u, 1024, xpw, nullptr, nullptr, dbc, BLROWS, 64, 1024);

    // 4. dt_proj + softplus: delta              [16384,1024] K=32 (A = dbc cols 0..31, lda=64)
    gemm_wmma_kernel<2><<<dim3(1024 / 64, gy), blk, 0, stream>>>(
        dbc, 64, dtw, dtb, nullptr, delta, BLROWS, 1024, 32);

    // 5. selective scan (+ Dp skip, silu(z) gate) -> y
    scan_kernel<<<(BATCH * D_INNER + 255) / 256, blk, 0, stream>>>(
        u, delta, dbc, xz, alog, Dp, y);

    // 6. out_proj: h = y @ out_proj_w           [16384,512] K=1024
    gemm_wmma_kernel<0><<<dim3(512 / 64, gy), blk, 0, stream>>>(
        y, 1024, opw, nullptr, nullptr, h, BLROWS, 512, 1024);

    // 7. LayerNorm -> h2
    ln_kernel<<<BLROWS, blk, 0, stream>>>(h, lng, lnb, h2);

    // 8. lin + bias + erf-GELU + residual -> out
    gemm_wmma_kernel<3><<<dim3(512 / 64, gy), blk, 0, stream>>>(
        h2, 512, linw, linb, x, out, BLROWS, 512, 512);
}
